// Net_18021682774696
// MI455X (gfx1250) — matrix operand order repaired and checked by reading, not compiled, not run on hardware
//
#include <hip/hip_runtime.h>
#include <hip/hip_bf16.h>

// ---------------- problem constants ----------------
#define BB 8
#define NN 2048
#define EE 32
#define HH 128
#define SS 256
#define TT 12
#define MROWS (BB * NN)          // 16384 node-rows, flattened
#define ALPHA_F 0.05f
#define LDT 40                   // LDS tile row stride in bf16 (80B = 5*16B, bank-friendly)

typedef __bf16 bf16_t;
typedef __attribute__((ext_vector_type(16))) __bf16 v16bf;
typedef __attribute__((ext_vector_type(8)))  float  v8f;

#define WMMA_BF16(A, Bf, C) \
  __builtin_amdgcn_wmma_f32_16x16x32_bf16(false, (A), false, (Bf), (short)0, (C), false, false)

// ---------------- WMMA fragment loaders (CDNA5 wave32 layouts) ----------------
// A fragment 16x32 bf16 from row-major LDS tile [m][k] (row stride LDT):
// lane: m = lane&15, hi = (lane>>4)*8 ; needs K = hi..hi+7 then 16+hi..16+hi+7
// => two contiguous 16B runs -> 2x ds_load_b128
__device__ __forceinline__ v16bf a_frag(const bf16_t* tile, int mbase, int lane) {
  const bf16_t* row = tile + (size_t)(mbase + (lane & 15)) * LDT + ((lane >> 4) << 3);
  union { uint4 u[2]; v16bf v; } u;
  u.u[0] = *(const uint4*)(row);
  u.u[1] = *(const uint4*)(row + 16);
  return u.v;
}

// B fragment 32x16 bf16 from TRANSPOSED LDS tile [n][k] (row stride LDT):
// lane: n = lane&15, hi = (lane>>4)*8 ; needs K = hi..hi+7 then 16+hi..16+hi+7
// => two contiguous 16B runs -> 2x ds_load_b128
__device__ __forceinline__ v16bf b_frag(const bf16_t* tile, int nbase, int lane) {
  const bf16_t* row = tile + (size_t)(nbase + (lane & 15)) * LDT + ((lane >> 4) << 3);
  union { uint4 u[2]; v16bf v; } u;
  u.u[0] = *(const uint4*)(row);
  u.u[1] = *(const uint4*)(row + 16);
  return u.v;
}

// cvt helper: write 4 floats as bf16 to LDS (contiguous, 8B-aligned)
__device__ __forceinline__ void st4_bf16(bf16_t* d, float4 v) {
  d[0] = (bf16_t)v.x; d[1] = (bf16_t)v.y; d[2] = (bf16_t)v.z; d[3] = (bf16_t)v.w;
}

// ---------------- adjacency normalization ----------------
__global__ void k_rowsum_norm_static(const float* __restrict__ A, bf16_t* __restrict__ An) {
  __shared__ float red[256];
  int v = blockIdx.x, tid = threadIdx.x;
  const float* row = A + (size_t)v * NN;
  float s = 0.f;
  for (int w = tid; w < NN; w += 256) s += row[w];
  red[tid] = s; __syncthreads();
  for (int o = 128; o > 0; o >>= 1) { if (tid < o) red[tid] += red[tid + o]; __syncthreads(); }
  float inv = 1.0f / (red[0] + 1.0f);
  for (int w = tid; w < NN; w += 256) {
    float val = row[w] + (w == v ? 1.f : 0.f);
    An[(size_t)v * NN + w] = (bf16_t)(val * inv);
  }
}

__global__ void k_rowsum_dyn(const float* __restrict__ A, float* __restrict__ rowsum) {
  __shared__ float red[256];
  int v = blockIdx.x, b = blockIdx.y, tid = threadIdx.x;
  const float* row = A + ((size_t)b * NN + v) * NN;
  float s = 0.f;
  for (int w = tid; w < NN; w += 256) s += row[w];
  red[tid] = s; __syncthreads();
  for (int o = 128; o > 0; o >>= 1) { if (tid < o) red[tid] += red[tid + o]; __syncthreads(); }
  if (tid == 0) rowsum[b * NN + v] = red[0] + 1.0f;
}

__global__ void k_colsum_dyn(const float* __restrict__ A, float* __restrict__ colsum) {
  int b = blockIdx.y;
  int w = blockIdx.x * 256 + threadIdx.x;
  const float* Ab = A + (size_t)b * NN * NN;
  float s = 0.f;
  for (int v = 0; v < NN; ++v) s += Ab[(size_t)v * NN + w];
  colsum[b * NN + w] = s + 1.0f;
}

// writes row-normalized A (bf16, coalesced) and row-normalized A^T (bf16, via LDS transpose)
__global__ void k_norm_dyn_tr(const float* __restrict__ A, const float* __restrict__ rowsum,
                              const float* __restrict__ colsum,
                              bf16_t* __restrict__ An, bf16_t* __restrict__ ATn) {
  __shared__ float tile[32][33];
  int b = blockIdx.z;
  int w0 = blockIdx.x * 32, v0 = blockIdx.y * 32;
  int tx = threadIdx.x, ty = threadIdx.y;   // 32 x 8
  const float* Ab = A + (size_t)b * NN * NN;
#pragma unroll
  for (int t = 0; t < 4; ++t) {
    int v = v0 + ty + 8 * t;
    int w = w0 + tx;
    float raw = Ab[(size_t)v * NN + w] + (v == w ? 1.f : 0.f);
    An[(size_t)b * NN * NN + (size_t)v * NN + w] = (bf16_t)(raw / rowsum[b * NN + v]);
    tile[ty + 8 * t][tx] = raw / colsum[b * NN + w];   // destined for A^T[w][v]
  }
  __syncthreads();
#pragma unroll
  for (int t = 0; t < 4; ++t) {
    int wr = w0 + ty + 8 * t;
    int vc = v0 + tx;
    ATn[(size_t)b * NN * NN + (size_t)wr * NN + vc] = (bf16_t)tile[tx][ty + 8 * t];
  }
}

// ---------------- layout / elementwise helpers ----------------
__global__ void k_chan_to_nm(const float* __restrict__ in, float* __restrict__ out) {
  int idx = blockIdx.x * 256 + threadIdx.x;        // over B*N*32
  int c = idx & 31; int bn = idx >> 5; int b = bn >> 11; int n = bn & (NN - 1);
  out[idx] = in[((size_t)b * EE + c) * NN + n];
}

__global__ void k_build_base(const float* __restrict__ sp, const float* __restrict__ td,
                             const float* __restrict__ tw, float* __restrict__ base) {
  int idx = blockIdx.x * 256 + threadIdx.x;        // over B*N*128
  int c = idx & 127; int bn = idx >> 7; int b = bn >> 11; int n = bn & (NN - 1);
  float v = 0.f;
  if (c >= 96)      v = tw[((size_t)b * 32 + (c - 96)) * NN + n];
  else if (c >= 64) v = td[((size_t)b * 32 + (c - 64)) * NN + n];
  else if (c >= 32) v = sp[((size_t)b * 32 + (c - 32)) * NN + n];
  base[idx] = v;
}

__global__ void k_make_hidden(const float* __restrict__ x_nm, const float* __restrict__ base,
                              float* __restrict__ hid) {
  int idx = blockIdx.x * 256 + threadIdx.x;        // over B*N*128
  int c = idx & 127; int bn = idx >> 7;
  hid[idx] = (c < 32) ? x_nm[(size_t)bn * EE + c] : base[idx];
}

__global__ void k_copy(const float* __restrict__ in, float* __restrict__ out) {
  int idx = blockIdx.x * 256 + threadIdx.x;
  out[idx] = in[idx];
}

// copy x (rows x 32) into cols 0..31 of the 96-wide hop-concat buffer
__global__ void k_copy_x_ho(const float* __restrict__ x_nm, float* __restrict__ ho) {
  int idx = blockIdx.x * 256 + threadIdx.x;        // over B*N*32
  int c = idx & 31; int row = idx >> 5;
  ho[(size_t)row * 96 + c] = x_nm[idx];
}

__global__ void k_zero(float* __restrict__ out) {
  int idx = blockIdx.x * 256 + threadIdx.x;
  out[idx] = 0.f;
}

// ---------------- gated TCN GEMM: Out = tanh(Wf In + bf) * sigmoid(Wg In + bg) ------------
__global__ void k_gemm_gated(const float* __restrict__ In,
                             const float* __restrict__ Wf, const float* __restrict__ biasF,
                             const float* __restrict__ Wg, const float* __restrict__ biasG,
                             float* __restrict__ Out) {
  __shared__ __align__(16) bf16_t inT[64 * LDT];
  __shared__ __align__(16) bf16_t wfT[32 * LDT];   // transposed: [o][k]
  __shared__ __align__(16) bf16_t wgT[32 * LDT];
  int tid = threadIdx.x, lane = tid & 31, wv = tid >> 5;
  int row0 = blockIdx.x * 64, o0 = blockIdx.y * 32;
  v8f aF0 = {}, aF1 = {}, aG0 = {}, aG1 = {};
  for (int kc = 0; kc < HH; kc += 32) {
    // ---- load phase: distinct temporaries so loads pipeline ----
    float4 tIn[4], tWf[2], tWg[2];
#pragma unroll
    for (int i = 0; i < 4; ++i) {                  // 64x32 floats = 512 float4
      int idx = tid + i * 128; int r = idx >> 3, c4 = (idx & 7) * 4;
      tIn[i] = *(const float4*)(In + (size_t)(row0 + r) * HH + kc + c4);
    }
#pragma unroll
    for (int i = 0; i < 2; ++i) {
      int idx = tid + i * 128; int o = idx >> 3, c4 = (idx & 7) * 4;
      tWf[i] = *(const float4*)(Wf + (size_t)(o0 + o) * HH + kc + c4);
      tWg[i] = *(const float4*)(Wg + (size_t)(o0 + o) * HH + kc + c4);
    }
    // ---- convert/store phase ----
#pragma unroll
    for (int i = 0; i < 4; ++i) {
      int idx = tid + i * 128; int r = idx >> 3, c4 = (idx & 7) * 4;
      st4_bf16(inT + r * LDT + c4, tIn[i]);
    }
#pragma unroll
    for (int i = 0; i < 2; ++i) {
      int idx = tid + i * 128; int o = idx >> 3, c4 = (idx & 7) * 4;
      st4_bf16(wfT + o * LDT + c4, tWf[i]);
      st4_bf16(wgT + o * LDT + c4, tWg[i]);
    }
    __syncthreads();
    v16bf a   = a_frag(inT, wv * 16, lane);
    v16bf bF0 = b_frag(wfT, 0, lane);
    v16bf bF1 = b_frag(wfT, 16, lane);
    v16bf bG0 = b_frag(wgT, 0, lane);
    v16bf bG1 = b_frag(wgT, 16, lane);
    aF0 = WMMA_BF16(a, bF0, aF0);
    aF1 = WMMA_BF16(a, bF1, aF1);
    aG0 = WMMA_BF16(a, bG0, aG0);
    aG1 = WMMA_BF16(a, bG1, aG1);
    __syncthreads();
  }
  int mrow = row0 + wv * 16 + (lane >> 4) * 8;
  int n0 = o0 + (lane & 15);
  float bf0 = biasF[n0], bf1 = biasF[n0 + 16];
  float bg0 = biasG[n0], bg1 = biasG[n0 + 16];
#pragma unroll
  for (int r = 0; r < 8; ++r) {
    size_t ro = (size_t)(mrow + r) * HH;
    float f0 = tanhf(aF0[r] + bf0);
    float g0 = 1.f / (1.f + __expf(-(aG0[r] + bg0)));
    Out[ro + n0] = f0 * g0;
    float f1 = tanhf(aF1[r] + bf1);
    float g1 = 1.f / (1.f + __expf(-(aG1[r] + bg1)));
    Out[ro + n0 + 16] = f1 * g1;
  }
}

// ------------- generic GEMM: Out[row,o] (+)= sum_k In[row,k] W[o,k] + bias[o] -------------
// In row stride == K (32/96/128).  W row stride == ldw.
__global__ void k_gemm_bias_acc(const float* __restrict__ In, const float* __restrict__ W,
                                int ldw, const float* __restrict__ bias,
                                float* __restrict__ Out, int ldo, int K, int accflag) {
  __shared__ __align__(16) bf16_t inT[64 * LDT];
  __shared__ __align__(16) bf16_t wT[32 * LDT];    // transposed: [o][k]
  int tid = threadIdx.x, lane = tid & 31, wv = tid >> 5;
  int row0 = blockIdx.x * 64, o0 = blockIdx.y * 32;
  v8f a0 = {}, a1 = {};
  for (int kc = 0; kc < K; kc += 32) {
    float4 tIn[4], tW[2];
#pragma unroll
    for (int i = 0; i < 4; ++i) {
      int idx = tid + i * 128; int r = idx >> 3, c4 = (idx & 7) * 4;
      tIn[i] = *(const float4*)(In + (size_t)(row0 + r) * K + kc + c4);
    }
#pragma unroll
    for (int i = 0; i < 2; ++i) {
      int idx = tid + i * 128; int o = idx >> 3, c4 = (idx & 7) * 4;
      tW[i] = *(const float4*)(W + (size_t)(o0 + o) * ldw + kc + c4);
    }
#pragma unroll
    for (int i = 0; i < 4; ++i) {
      int idx = tid + i * 128; int r = idx >> 3, c4 = (idx & 7) * 4;
      st4_bf16(inT + r * LDT + c4, tIn[i]);
    }
#pragma unroll
    for (int i = 0; i < 2; ++i) {
      int idx = tid + i * 128; int o = idx >> 3, c4 = (idx & 7) * 4;
      st4_bf16(wT + o * LDT + c4, tW[i]);
    }
    __syncthreads();
    v16bf a  = a_frag(inT, wv * 16, lane);
    v16bf b0 = b_frag(wT, 0, lane);
    v16bf b1 = b_frag(wT, 16, lane);
    a0 = WMMA_BF16(a, b0, a0);
    a1 = WMMA_BF16(a, b1, a1);
    __syncthreads();
  }
  int mrow = row0 + wv * 16 + (lane >> 4) * 8;
  int n0 = o0 + (lane & 15);
  float bv0 = bias ? bias[n0] : 0.f;
  float bv1 = bias ? bias[n0 + 16] : 0.f;
#pragma unroll
  for (int r = 0; r < 8; ++r) {
    size_t ro = (size_t)(mrow + r) * ldo;
    float v0 = a0[r] + bv0, v1 = a1[r] + bv1;
    if (accflag) { Out[ro + n0] += v0; Out[ro + n0 + 16] += v1; }
    else         { Out[ro + n0]  = v0; Out[ro + n0 + 16]  = v1; }
  }
}

// ---- graph propagation: Hout[b,v,:] = alpha*X[b,v,:] + (1-alpha)*sum_w A[v,w] Hsrc[b,w,:] ----
// Hsrc has row stride ldh; Hout row stride ldo (column offset folded into pointer).
__global__ void k_gemm_prop(const bf16_t* __restrict__ A, long long astride,
                            const float* __restrict__ Hsrc, int ldh,
                            const float* __restrict__ X,
                            float* __restrict__ Hout, int ldo) {
  __shared__ __align__(16) bf16_t aT[64 * LDT];
  __shared__ __align__(16) bf16_t hT[32 * LDT];    // transposed: [c][k]
  int tid = threadIdx.x, lane = tid & 31, wv = tid >> 5;
  int b = blockIdx.y;
  int v0 = blockIdx.x * 64;
  const bf16_t* Ab = A + (size_t)b * (size_t)astride;
  size_t hbase = (size_t)b * NN * ldh;
  size_t xbase = (size_t)b * NN * EE;
  size_t obase = (size_t)b * NN * ldo;
  v8f a0 = {}, a1 = {};
  for (int kc = 0; kc < NN; kc += 32) {
    // ---- load phase: distinct temporaries so loads pipeline ----
    uint4 tA[2]; float4 tH[2];
#pragma unroll
    for (int i = 0; i < 2; ++i) {
      int idx = tid + i * 128; int r = idx >> 2, c4 = (idx & 3) * 8;
      tA[i] = *(const uint4*)(Ab + (size_t)(v0 + r) * NN + kc + c4);
    }
#pragma unroll
    for (int i = 0; i < 2; ++i) {
      int idx = tid + i * 128; int k = idx >> 3, c4 = (idx & 7) * 4;
      tH[i] = *(const float4*)(Hsrc + hbase + (size_t)(kc + k) * ldh + c4);
    }
    if (kc + 32 < NN)
      __builtin_prefetch(Ab + (size_t)(v0 + (tid >> 1)) * NN + kc + 32, 0, 1);
    // ---- convert/store phase ----
#pragma unroll
    for (int i = 0; i < 2; ++i) {
      int idx = tid + i * 128; int r = idx >> 2, c4 = (idx & 3) * 8;
      *(uint4*)(aT + r * LDT + c4) = tA[i];
    }
#pragma unroll
    for (int i = 0; i < 2; ++i) {
      int idx = tid + i * 128; int k = idx >> 3, c4 = (idx & 7) * 4;
      hT[(c4 + 0) * LDT + k] = (bf16_t)tH[i].x;
      hT[(c4 + 1) * LDT + k] = (bf16_t)tH[i].y;
      hT[(c4 + 2) * LDT + k] = (bf16_t)tH[i].z;
      hT[(c4 + 3) * LDT + k] = (bf16_t)tH[i].w;
    }
    __syncthreads();
    v16bf a  = a_frag(aT, wv * 16, lane);
    v16bf b0 = b_frag(hT, 0, lane);
    v16bf b1 = b_frag(hT, 16, lane);
    a0 = WMMA_BF16(a, b0, a0);
    a1 = WMMA_BF16(a, b1, a1);
    __syncthreads();
  }
  int vrow = v0 + wv * 16 + (lane >> 4) * 8;
  int c0 = lane & 15;
#pragma unroll
  for (int r = 0; r < 8; ++r) {
    size_t xo = xbase + (size_t)(vrow + r) * EE;
    size_t oo = obase + (size_t)(vrow + r) * ldo;
    Hout[oo + c0]      = ALPHA_F * X[xo + c0]      + (1.f - ALPHA_F) * a0[r];
    Hout[oo + c0 + 16] = ALPHA_F * X[xo + c0 + 16] + (1.f - ALPHA_F) * a1[r];
  }
}

// ---------------- layernorm over (C,N) per batch + relu ----------------
__global__ void k_ln_stats(const float* __restrict__ x, float* __restrict__ mu,
                           float* __restrict__ rstd) {
  __shared__ float r1[256], r2[256];
  int b = blockIdx.x, tid = threadIdx.x;
  const float* xb = x + (size_t)b * NN * EE;
  float s1 = 0.f, s2 = 0.f;
  for (int i = tid; i < NN * EE; i += 256) { float v = xb[i]; s1 += v; s2 += v * v; }
  r1[tid] = s1; r2[tid] = s2; __syncthreads();
  for (int o = 128; o > 0; o >>= 1) {
    if (tid < o) { r1[tid] += r1[tid + o]; r2[tid] += r2[tid + o]; }
    __syncthreads();
  }
  if (tid == 0) {
    const float inv = 1.0f / (float)(NN * EE);
    float m = r1[0] * inv;
    float var = r2[0] * inv - m * m;
    mu[b] = m;
    rstd[b] = rsqrtf(var + 1e-5f);
  }
}

__global__ void k_ln_apply(const float* __restrict__ xin, const float* __restrict__ mu,
                           const float* __restrict__ rstd, const float* __restrict__ w,
                           const float* __restrict__ bvec, float* __restrict__ out) {
  int idx = blockIdx.x * 256 + threadIdx.x;        // over B*N*32
  int c = idx & 31; int bn = idx >> 5; int b = bn >> 11; int n = bn & (NN - 1);
  float v = (xin[idx] - mu[b]) * rstd[b] * w[(size_t)c * NN + n] + bvec[(size_t)c * NN + n];
  out[idx] = fmaxf(v, 0.f);
}

// ---------------- end conv: out[b,t,n] = end_W @ skip + end_b (O=12, padded to 16) ----------
__global__ void k_end(const float* __restrict__ skip, const float* __restrict__ W,
                      const float* __restrict__ bias, float* __restrict__ out) {
  __shared__ __align__(16) bf16_t inT[64 * LDT];
  __shared__ __align__(16) bf16_t wT[32 * LDT];    // transposed: [o][k]
  int tid = threadIdx.x, lane = tid & 31, wv = tid >> 5;
  int row0 = blockIdx.x * 64;
  v8f acc = {};
  for (int kc = 0; kc < SS; kc += 32) {
    float4 tIn[4], tW[2];
#pragma unroll
    for (int i = 0; i < 4; ++i) {
      int idx = tid + i * 128; int r = idx >> 3, c4 = (idx & 7) * 4;
      tIn[i] = *(const float4*)(skip + (size_t)(row0 + r) * SS + kc + c4);
    }
#pragma unroll
    for (int i = 0; i < 2; ++i) {
      int idx = tid + i * 128; int o = idx >> 3, c4 = (idx & 7) * 4;
      tW[i] = (o < TT) ? *(const float4*)(W + (size_t)o * SS + kc + c4)
                       : make_float4(0.f, 0.f, 0.f, 0.f);
    }
#pragma unroll
    for (int i = 0; i < 4; ++i) {
      int idx = tid + i * 128; int r = idx >> 3, c4 = (idx & 7) * 4;
      st4_bf16(inT + r * LDT + c4, tIn[i]);
    }
#pragma unroll
    for (int i = 0; i < 2; ++i) {
      int idx = tid + i * 128; int o = idx >> 3, c4 = (idx & 7) * 4;
      st4_bf16(wT + o * LDT + c4, tW[i]);
    }
    __syncthreads();
    v16bf a  = a_frag(inT, wv * 16, lane);
    v16bf b0 = b_frag(wT, 0, lane);
    acc = WMMA_BF16(a, b0, acc);
    __syncthreads();
  }
  int col = lane & 15;
  if (col < TT) {
    int mrow = row0 + wv * 16 + (lane >> 4) * 8;
    float bv = bias[col];
#pragma unroll
    for (int r = 0; r < 8; ++r) {
      int row = mrow + r;
      int b = row >> 11, n = row & (NN - 1);
      out[((size_t)b * TT + col) * NN + n] = acc[r] + bv;
    }
  }
}

// ---------------- host orchestration ----------------
extern "C" void kernel_launch(void* const* d_in, const int* in_sizes, int n_in,
                              void* d_out, int out_size, void* d_ws, size_t ws_size,
                              hipStream_t stream) {
  (void)in_sizes; (void)n_in; (void)out_size; (void)ws_size;

  const float* x       = (const float*)d_in[0];
  const float* dy      = (const float*)d_in[1];
  const float* statg   = (const float*)d_in[2];
  const float* sp      = (const float*)d_in[3];
  const float* td      = (const float*)d_in[4];
  const float* tw      = (const float*)d_in[5];
  const float* enc_Wf  = (const float*)d_in[6];
  const float* enc_bf  = (const float*)d_in[7];
  const float* enc_Wg  = (const float*)d_in[8];
  const float* enc_bg  = (const float*)d_in[9];
  const float* skip_W  = (const float*)d_in[10];
  const float* skip_b  = (const float*)d_in[11];
  const float* norm_w  = (const float*)d_in[12];
  const float* norm_b  = (const float*)d_in[13];
  const float* gW[3]   = {(const float*)d_in[14], (const float*)d_in[16], (const float*)d_in[18]};
  const float* gb[3]   = {(const float*)d_in[15], (const float*)d_in[17], (const float*)d_in[19]};
  const float* skipE_W = (const float*)d_in[20];
  const float* skipE_b = (const float*)d_in[21];
  const float* end_W   = (const float*)d_in[22];
  const float* end_b   = (const float*)d_in[23];
  float* out = (float*)d_out;

  // bump allocator over workspace (~190 MB used)
  char* p = (char*)d_ws;
  auto alloc = [&](size_t bytes) -> void* {
    void* q = (void*)p;
    p += (bytes + 255) & ~(size_t)255;
    return q;
  };
  bf16_t* asn    = (bf16_t*)alloc((size_t)NN * NN * 2);            //  8 MB static norm
  bf16_t* adn    = (bf16_t*)alloc((size_t)BB * NN * NN * 2);       // 64 MB dyn norm
  bf16_t* adtn   = (bf16_t*)alloc((size_t)BB * NN * NN * 2);       // 64 MB dyn^T norm
  float*  rowsum = (float*)alloc((size_t)BB * NN * 4);
  float*  colsum = (float*)alloc((size_t)BB * NN * 4);
  float*  x_nm   = (float*)alloc((size_t)MROWS * EE * 4);
  float*  x_new  = (float*)alloc((size_t)MROWS * EE * 4);
  float*  base   = (float*)alloc((size_t)MROWS * HH * 4);
  float*  hidA   = (float*)alloc((size_t)MROWS * HH * 4);
  float*  hidB   = (float*)alloc((size_t)MROWS * HH * 4);
  float*  skip   = (float*)alloc((size_t)MROWS * SS * 4);
  float*  ho     = (float*)alloc((size_t)MROWS * 96 * 4);          // hop concat [x|h1|h2]
  float*  mu     = (float*)alloc((size_t)BB * 4);
  float*  rstd   = (float*)alloc((size_t)BB * 4);

  // --- one-time adjacency normalization (bf16; all 3 graphs ~136MB fit in 192MB L2) ---
  k_rowsum_norm_static<<<NN, 256, 0, stream>>>(statg, asn);
  k_rowsum_dyn<<<dim3(NN, BB), 256, 0, stream>>>(dy, rowsum);
  k_colsum_dyn<<<dim3(NN / 256, BB), 256, 0, stream>>>(dy, colsum);
  k_norm_dyn_tr<<<dim3(NN / 32, NN / 32, BB), dim3(32, 8), 0, stream>>>(dy, rowsum, colsum, adn, adtn);

  k_chan_to_nm<<<MROWS * EE / 256, 256, 0, stream>>>(x, x_nm);
  k_build_base<<<MROWS * HH / 256, 256, 0, stream>>>(sp, td, tw, base);
  k_zero<<<MROWS * SS / 256, 256, 0, stream>>>(skip);

  for (int i = 0; i < 3; ++i) {
    // hidden = concat(x, embeddings) ; two gated TCN units (ping-pong)
    k_make_hidden<<<MROWS * HH / 256, 256, 0, stream>>>(x_nm, base, hidA);
    k_gemm_gated<<<dim3(MROWS / 64, HH / 32), 128, 0, stream>>>(
        hidA, enc_Wf + ((size_t)i * 2 + 0) * HH * HH, enc_bf + ((size_t)i * 2 + 0) * HH,
              enc_Wg + ((size_t)i * 2 + 0) * HH * HH, enc_bg + ((size_t)i * 2 + 0) * HH, hidB);
    k_gemm_gated<<<dim3(MROWS / 64, HH / 32), 128, 0, stream>>>(
        hidB, enc_Wf + ((size_t)i * 2 + 1) * HH * HH, enc_bf + ((size_t)i * 2 + 1) * HH,
              enc_Wg + ((size_t)i * 2 + 1) * HH * HH, enc_bg + ((size_t)i * 2 + 1) * HH, hidA);

    // skip += skip_W[i] @ hidden + skip_b[i]
    k_gemm_bias_acc<<<dim3(MROWS / 64, SS / 32), 128, 0, stream>>>(
        hidA, skip_W + (size_t)i * SS * HH, HH, skip_b + (size_t)i * SS, skip, SS, HH, 1);

    // x_new starts as the residual; ho cols 0..31 hold hop-0 (= x) for all graphs
    k_copy<<<MROWS * EE / 256, 256, 0, stream>>>(x_nm, x_new);
    k_copy_x_ho<<<MROWS * EE / 256, 256, 0, stream>>>(x_nm, ho);
    for (int g = 0; g < 3; ++g) {
      const bf16_t* Ag = (g == 0) ? asn : (g == 1) ? adn : adtn;
      long long stride = (g == 0) ? 0 : (long long)NN * NN;
      // hop1 -> ho cols 32..63 ; hop2 -> ho cols 64..95
      k_gemm_prop<<<dim3(NN / 64, BB), 128, 0, stream>>>(Ag, stride, x_nm, EE, x_nm, ho + 32, 96);
      k_gemm_prop<<<dim3(NN / 64, BB), 128, 0, stream>>>(Ag, stride, ho + 32, 96, x_nm, ho + 64, 96);
      // x_new += gW[g][i] @ concat(x,h1,h2) + gb[g][i]   (single K=96 GEMM)
      k_gemm_bias_acc<<<dim3(MROWS / 64, 1), 128, 0, stream>>>(
          ho, gW[g] + (size_t)i * EE * 96, 96, gb[g] + (size_t)i * EE, x_new, EE, 96, 1);
    }

    // layernorm over (C,N) per batch + relu  ->  x for next layer
    k_ln_stats<<<BB, 256, 0, stream>>>(x_new, mu, rstd);
    k_ln_apply<<<MROWS * EE / 256, 256, 0, stream>>>(
        x_new, mu, rstd, norm_w + (size_t)i * EE * NN, norm_b + (size_t)i * EE * NN, x_nm);
  }

  // skip += skipE_W @ x + skipE_b ; out = end_W @ skip + end_b
  k_gemm_bias_acc<<<dim3(MROWS / 64, SS / 32), 128, 0, stream>>>(
      x_nm, skipE_W, EE, skipE_b, skip, SS, EE, 1);
  k_end<<<MROWS / 64, 128, 0, stream>>>(skip, end_W, end_b, out);
}
